// DynamicLoop_31190052504238
// MI455X (gfx1250) — matrix operand order repaired and checked by reading, not compiled, not run on hardware
//
#include <hip/hip_runtime.h>

// ---------------------------------------------------------------------------
// CDNA5 (gfx1250) recurrent wave-sim.
// - 3x3 conv (42->32) per step as implicit GEMM on v_wmma_f32_16x16x32_f16.
// - Source tensor pixel-major f16 in a zero-border-padded global buffer so the
//   halo tile is a regular 2-D tile: staged into LDS by the Tensor Data Mover
//   (tensor_load_to_lds + s_wait_tensorcnt) issued by wave 0, while waves 1..7
//   stage the reordered weights + small params.
// - fp32 kept for the complex state s_r/s_i and stat accumulators.
// ---------------------------------------------------------------------------

#define BATCH   8
#define IMH     256
#define IMW     256
#define NPIX    (IMH * IMW)
#define PROWS   (IMH + 2)          // padded rows per batch image
#define PCOLS   (IMW + 2)          // padded cols
#define PPIX    (PROWS * PCOLS)
#define WD      32                 // hh channels / conv output channels
#define NSRC    42                 // real source channels
#define CPP     48                 // padded channels per pixel (per tap block)
#define TH      16
#define TW      16
#define LH      (TH + 2)
#define LW      (TW + 2)
#define KREAL   (9 * CPP)          // 432
#define KPAD    448                // 14 chunks of 32
#define KCHUNKS (KPAD / 32)        // 14
#define NSTEPS  16
#define DELAY_T 8

typedef __attribute__((ext_vector_type(16))) _Float16 v16h;
typedef __attribute__((ext_vector_type(8)))  float    v8f;
typedef __attribute__((ext_vector_type(4)))  unsigned int u32x4;
typedef __attribute__((ext_vector_type(4)))  int      i32x4;
typedef __attribute__((ext_vector_type(8)))  int      i32x8;

union V16H { uint4 u[2]; v16h v; };
union H8   { uint4 u; _Float16 h[8]; };

// accum layout (floats): [0..15] pre_wall, [16..31] delta, [32..48] norm,
// [49] sat, [50] leak, [52] cos_a, [53] sin_a, [54] decay
#define ACC_PW    0
#define ACC_DELTA 16
#define ACC_NORM  32
#define ACC_SAT   49
#define ACC_LEAK  50
#define ACC_COS   52
#define ACC_SIN   53
#define ACC_DEC   54

// shared memory byte layout for step kernel
#define SH_SRC_B  (LH * LW * CPP * 2)          // 31104 (= 18 rows x 1728 B)
#define SH_W_B    (WD * KPAD * 2)              // 28672
#define SH_GATE_B (48 * 4)
#define SH_BUP_B  (32 * 4)
#define SH_RED_B  (8 * 4)
#define SH_TOTAL  (SH_SRC_B + SH_W_B + SH_GATE_B + SH_BUP_B + SH_RED_B)

__device__ __forceinline__ float sigm(float x) { return 1.0f / (1.0f + __expf(-x)); }

// ---------------------------------------------------------------------------
// prep: reorder conv weights (32,42,3,3) f32 -> W16[m][tap*48+c] f16,
//       zero stat accumulators, precompute uniform rotation/decay params.
// ---------------------------------------------------------------------------
__global__ __launch_bounds__(256)
void prep_kernel(const float* __restrict__ w_up,
                 const float* __restrict__ angle_p, const float* __restrict__ decay_p,
                 _Float16* __restrict__ W16g, float* __restrict__ accum)
{
    int i = blockIdx.x * 256 + threadIdx.x;
    if (i < 52) accum[i] = 0.0f;
    if (i == ACC_COS || i == ACC_SIN) {
        float ang = tanhf(angle_p[0]) * 0.02f;
        accum[i] = (i == ACC_COS) ? __cosf(ang) : __sinf(ang);
    }
    if (i == ACC_DEC) accum[i] = 0.95f + 0.05f * sigm(decay_p[0]);
    if (i < WD * KPAD) {
        int m = i / KPAD, k = i - m * KPAD;
        int tap = k / CPP, c = k - tap * CPP;
        _Float16 v = (_Float16)0.0f;
        if (tap < 9 && c < NSRC) {
            int ky = tap / 3, kx = tap - ky * 3;
            v = (_Float16)w_up[((m * NSRC + c) * 3 + ky) * 3 + kx];
        }
        W16g[i] = v;
    }
}

// ---------------------------------------------------------------------------
// border: zero the 1-pixel borders of both ping-pong src buffers (they encode
// the conv/stencil zero padding and are never written by the step kernels)
// ---------------------------------------------------------------------------
__global__ __launch_bounds__(256)
void border_kernel(_Float16* __restrict__ srcA, _Float16* __restrict__ srcB)
{
    size_t gid = (size_t)blockIdx.x * 256 + threadIdx.x;
    if (gid >= (size_t)BATCH * PPIX) return;
    int rc = (int)(gid % PPIX);
    int r = rc / PCOLS, c = rc - r * PCOLS;
    if (r == 0 || r == PROWS - 1 || c == 0 || c == PCOLS - 1) {
        uint4 z = make_uint4(0u, 0u, 0u, 0u);
        uint4* pa = (uint4*)(srcA + gid * CPP);
        uint4* pb = (uint4*)(srcB + gid * CPP);
#pragma unroll
        for (int q = 0; q < CPP / 8; ++q) { pa[q] = z; pb[q] = z; }
    }
}

// ---------------------------------------------------------------------------
// init: build state, h = tanh(conv1x1) * free, write padded pixel-major f16 src
// ---------------------------------------------------------------------------
__global__ __launch_bounds__(256)
void init_kernel(const float* __restrict__ x, const float* __restrict__ w_in,
                 const float* __restrict__ b_in, _Float16* __restrict__ srcA,
                 float* __restrict__ sRa, float* __restrict__ sIa,
                 float* __restrict__ accum)
{
    __shared__ float red;
    if (threadIdx.x == 0) red = 0.0f;
    __syncthreads();

    size_t gid = (size_t)blockIdx.x * 256 + threadIdx.x;  // b*NPIX + p
    int b = (int)(gid / NPIX);
    int p = (int)(gid - (size_t)b * NPIX);
    int y = p / IMW, xx = p - y * IMW;

    const float* xb = x + (size_t)b * 6 * NPIX + p;
    float xc[6];
#pragma unroll
    for (int c = 0; c < 6; ++c) xc[c] = xb[(size_t)c * NPIX];

    float wall = xc[0], fre = 1.0f - wall;
    float sr = xc[1] * fre, si = xc[2] * fre;

    float in10[10] = { xc[0], xc[1], xc[2], xc[3], xc[4], xc[5], sr, si, sr, si };

    _Float16 buf[CPP];
#pragma unroll
    for (int o = 0; o < WD; ++o) {
        float acc = b_in[o];
#pragma unroll
        for (int c = 0; c < 10; ++c) acc += w_in[o * 10 + c] * in10[c];
        buf[o] = (_Float16)(tanhf(acc) * fre);
    }
#pragma unroll
    for (int c = 0; c < 6; ++c) buf[32 + c] = (_Float16)xc[c];
    buf[38] = (_Float16)sr;  buf[39] = (_Float16)si;
    buf[40] = (_Float16)sr;  buf[41] = (_Float16)si;   // frontier = state
#pragma unroll
    for (int c = 42; c < CPP; ++c) buf[c] = (_Float16)0.0f;

    size_t ppad = ((size_t)b * PROWS + y + 1) * PCOLS + xx + 1;
    uint4* dst = (uint4*)(srcA + ppad * CPP);
    const uint4* sb = (const uint4*)buf;
#pragma unroll
    for (int q = 0; q < CPP / 8; ++q) dst[q] = sb[q];

    sRa[gid] = sr;
    sIa[gid] = si;

    atomicAdd(&red, sqrtf(sr * sr + si * si));
    __syncthreads();
    if (threadIdx.x == 0) atomicAdd(&accum[ACC_NORM + 0], red);
}

// ---------------------------------------------------------------------------
// step: one recurrence step.
// ---------------------------------------------------------------------------
__global__ __launch_bounds__(256)
void step_kernel(const _Float16* __restrict__ srcCur, _Float16* __restrict__ srcNext,
                 const float* __restrict__ sRcur, const float* __restrict__ sIcur,
                 float* __restrict__ sRnext, float* __restrict__ sInext,
                 const _Float16* __restrict__ W16g,
                 const float* __restrict__ w_gate, const float* __restrict__ b_gate,
                 const float* __restrict__ b_up,
                 float* __restrict__ accum, int step)
{
    const int tid = threadIdx.x;
    const int bx = blockIdx.x, by = blockIdx.y, b = blockIdx.z;

    extern __shared__ char smem[];
    _Float16* sSrc  = (_Float16*)smem;                       // [LH][LW][CPP]
    _Float16* sW    = (_Float16*)(smem + SH_SRC_B);          // [WD][KPAD]
    float*    sGate = (float*)(smem + SH_SRC_B + SH_W_B);
    float*    sBup  = (float*)(smem + SH_SRC_B + SH_W_B + SH_GATE_B);
    float*    sRed  = (float*)(smem + SH_SRC_B + SH_W_B + SH_GATE_B + SH_BUP_B);

    // tile start in the padded image: (by*TH, bx*TW), 18x18 pixels, always valid
    const _Float16* tileG =
        srcCur + (((size_t)b * PROWS + by * TH) * PCOLS + bx * TW) * CPP;

#if __has_builtin(__builtin_amdgcn_tensor_load_to_lds)
    if (tid < 32) {
        // ---- Tensor Data Mover: 2-D tile 18 rows x 1728 B (216 x 8B elems),
        // row stride PCOLS*96 B (3096 x 8B elems) -> LDS offset 0.
        unsigned long long ga = (unsigned long long)tileG;
        const unsigned TD0 = 4096u, TD1 = 4096u;       // never clipped
        const unsigned TILE0 = (LW * CPP * 2) / 8;     // 216
        const unsigned STR0 = (PCOLS * CPP * 2) / 8;   // 3096
        u32x4 g0 = { 1u,                               // count=1, load, user D#
                     0u,                               // lds_addr = 0 (sSrc base)
                     (unsigned)(ga & 0xFFFFFFFFu),
                     (unsigned)((ga >> 32) & 0x1FFFFFFu) | (2u << 30) }; // type=2
        i32x8 g1 = { (int)(3u << 16),                  // data_size=8B, mask=0
                     (int)((TD0 & 0xFFFFu) << 16),     // tensor_dim0 lo16
                     (int)((TD0 >> 16) | ((TD1 & 0xFFFFu) << 16)),
                     (int)((TD1 >> 16) | (TILE0 << 16)),
                     (int)LH,                          // tile_dim1=18, tile_dim2=0
                     (int)STR0,                        // dim0 stride lo32
                     0, 0 };
        i32x4 z4 = { 0, 0, 0, 0 };
#if defined(__clang_major__) && (__clang_major__ >= 23)
        i32x8 z8 = { 0, 0, 0, 0, 0, 0, 0, 0 };
        __builtin_amdgcn_tensor_load_to_lds(g0, g1, z4, z4, z8, 0);
#else
        __builtin_amdgcn_tensor_load_to_lds(g0, g1, z4, z4, 0);
#endif
        __builtin_amdgcn_s_wait_tensorcnt(0);
    } else {
        // waves 1..7 stage weights + small params in parallel with the TDM
        const int u = tid - 32;                        // 0..223
        const uint4* wsrc = (const uint4*)W16g;
        uint4* wdst = (uint4*)sW;
        for (int i = u; i < WD * KPAD / 8; i += 224) wdst[i] = wsrc[i];
        if (u < NSRC)                      sGate[u] = w_gate[u];
        else if (u < NSRC + WD)            sBup[u - NSRC] = b_up[u - NSRC];
        else if (u < NSRC + WD + 8)        sRed[u - NSRC - WD] = 0.0f;
    }
#else
    {   // fallback: manual strided fill (rows are 1728 B contiguous)
        const int ROWU4 = LW * CPP / 8;                // 108 uint4 per row
        for (int i = tid; i < LH * ROWU4; i += 256) {
            int r = i / ROWU4, q = i - r * ROWU4;
            *(uint4*)(sSrc + (size_t)r * LW * CPP + (size_t)q * 8) =
                *(const uint4*)(tileG + (size_t)r * PCOLS * CPP + (size_t)q * 8);
        }
        const uint4* wsrc = (const uint4*)W16g;
        uint4* wdst = (uint4*)sW;
        for (int i = tid; i < WD * KPAD / 8; i += 256) wdst[i] = wsrc[i];
        if (tid < NSRC) sGate[tid] = w_gate[tid];
        if (tid < WD)   sBup[tid]  = b_up[tid];
        if (tid < 8)    sRed[tid]  = 0.0f;
    }
#endif
    __syncthreads();

    // uniform params precomputed by prep_kernel (scalar loads)
    const float ca    = accum[ACC_COS];
    const float sa    = accum[ACC_SIN];
    const float decay = accum[ACC_DEC];

    // ---- pointwise wave physics: 1 pixel per thread -----------------------
    {
        int ly = tid >> 4, lx = tid & 15;
        const _Float16* pc = sSrc + ((size_t)(ly + 1) * LW + (lx + 1)) * CPP;
        const int xs = CPP, ys = LW * CPP;

        float wall = (float)pc[32];
        float fre  = 1.0f - wall;

        float fr = (float)pc[40 + xs] + (float)pc[40 - xs]
                 + (float)pc[40 + ys] + (float)pc[40 - ys];
        float fi = (float)pc[41 + xs] + (float)pc[41 - xs]
                 + (float)pc[41 + ys] + (float)pc[41 - ys];
        if (step == DELAY_T) { fr += (float)pc[35]; fi += (float)pc[36]; }

        float dot = b_gate[0];
#pragma unroll
        for (int c = 0; c < NSRC; ++c) dot += sGate[c] * (float)pc[c];
        float g = 0.998f + 0.002f * sigm(dot);
        fr *= g; fi *= g;

        float rr = (fr * ca - fi * sa) * fre;
        float ri = (fr * sa + fi * ca) * fre;

        float mag = sqrtf(rr * rr + ri * ri + 1e-8f);
        float sc  = fminf(2.0f / mag, 1.0f);
        rr *= sc; ri *= sc;

        float pwc = (fabsf(rr) + fabsf(ri)) * wall;

        size_t pidx = ((size_t)b * IMH + (by * TH + ly)) * IMW + (bx * TW + lx);
        float sr = sRcur[pidx], si = sIcur[pidx];
        float nr = decay * sr + rr, ni = decay * si + ri;
        mag = sqrtf(nr * nr + ni * ni + 1e-8f);
        sc  = fminf(2.0f / mag, 1.0f);
        nr = nr * sc * fre; ni = ni * sc * fre;
        sRnext[pidx] = nr;  sInext[pidx] = ni;

        float dsum = (nr - sr) * (nr - sr) + (ni - si) * (ni - si);
        float nrm  = sqrtf(nr * nr + ni * ni);
        atomicAdd(&sRed[0], pwc);
        atomicAdd(&sRed[1], dsum);
        atomicAdd(&sRed[2], nrm);
        if (step == NSTEPS) {
            float satc = (fabsf(nr) > 4.0f ? 1.0f : 0.0f) +
                         (fabsf(ni) > 4.0f ? 1.0f : 0.0f);
            float lkc  = (nrm > 0.5f && wall > 0.5f) ? 1.0f : 0.0f;
            atomicAdd(&sRed[3], satc);
            atomicAdd(&sRed[4], lkc);
        }

        // next-src channels 32..47: x copy, s, frontier(=inc), zero pad
        H8 c0, c1;
#pragma unroll
        for (int c = 0; c < 6; ++c) c0.h[c] = pc[32 + c];
        c0.h[6] = (_Float16)nr;  c0.h[7] = (_Float16)ni;
        c1.h[0] = (_Float16)rr;  c1.h[1] = (_Float16)ri;
#pragma unroll
        for (int c = 2; c < 8; ++c) c1.h[c] = (_Float16)0.0f;
        size_t ppad = ((size_t)b * PROWS + by * TH + ly + 1) * PCOLS + bx * TW + lx + 1;
        _Float16* outp = srcNext + ppad * CPP;
        *(uint4*)(outp + 32) = c0.u;
        *(uint4*)(outp + 40) = c1.u;
    }

    // ---- 3x3 conv 42->32 as implicit GEMM on WMMA -------------------------
    // Each wave: 2 row-strips of 16 px x 32 out channels.  A fragments are
    // shared across both strips (4 WMMA per A load pair).
    {
        const int wave = tid >> 5, lane = tid & 31;
        const int grp = lane >> 4, ln = lane & 15;     // wave32 half-groups
        const int row0 = wave * 2, row1 = row0 + 1;

        v8f acc00 = {}, acc10 = {}, acc01 = {}, acc11 = {};

#pragma unroll
        for (int kc = 0; kc < KCHUNKS; ++kc) {
            // A fragments (weights): K runs kc*32+8g+[0..7] and +16
            const int ka = kc * 32 + 8 * grp;
            const _Float16* ap0 = sW + (size_t)ln * KPAD + ka;
            const _Float16* ap1 = sW + (size_t)(16 + ln) * KPAD + ka;
            V16H a0, a1;
            a0.u[0] = *(const uint4*)ap0;  a0.u[1] = *(const uint4*)(ap0 + 16);
            a1.u[0] = *(const uint4*)ap1;  a1.u[1] = *(const uint4*)(ap1 + 16);

            // B fragments: K runs kc*32+8g+[0..7] and +16, each 8 contiguous
            // channels at one tap (aligned 8-runs never cross a 48-block)
            V16H b0, b1;
#pragma unroll
            for (int hh = 0; hh < 2; ++hh) {
                const int kb = kc * 32 + 8 * grp + 16 * hh;
                if (kb < KREAL) {
                    int tap = kb / CPP, c0 = kb - tap * CPP;
                    int ky = tap / 3, kx = tap - ky * 3;
                    const _Float16* bp0 =
                        sSrc + ((size_t)(row0 + ky) * LW + (ln + kx)) * CPP + c0;
                    const _Float16* bp1 =
                        sSrc + ((size_t)(row1 + ky) * LW + (ln + kx)) * CPP + c0;
                    b0.u[hh] = *(const uint4*)bp0;
                    b1.u[hh] = *(const uint4*)bp1;
                } else {
                    uint4 z = make_uint4(0u, 0u, 0u, 0u);
                    b0.u[hh] = z; b1.u[hh] = z;
                }
            }

            acc00 = __builtin_amdgcn_wmma_f32_16x16x32_f16(
                        false, a0.v, false, b0.v, (short)0, acc00, false, false);
            acc10 = __builtin_amdgcn_wmma_f32_16x16x32_f16(
                        false, a1.v, false, b0.v, (short)0, acc10, false, false);
            acc01 = __builtin_amdgcn_wmma_f32_16x16x32_f16(
                        false, a0.v, false, b1.v, (short)0, acc01, false, false);
            acc11 = __builtin_amdgcn_wmma_f32_16x16x32_f16(
                        false, a1.v, false, b1.v, (short)0, acc11, false, false);
        }

        // epilogue: lane owns pixel (row, ln); D VGPR v -> ch = Mtile*16+8g+v
#pragma unroll
        for (int sidx = 0; sidx < 2; ++sidx) {
            const int row = row0 + sidx;
            const _Float16* pc = sSrc + ((size_t)(row + 1) * LW + (ln + 1)) * CPP;
            float fre = 1.0f - (float)pc[32];
            size_t ppad = ((size_t)b * PROWS + by * TH + row + 1) * PCOLS
                        + bx * TW + ln + 1;
            _Float16* outp = srcNext + ppad * CPP;

#pragma unroll
            for (int t = 0; t < 2; ++t) {
                int base = t * 16 + 8 * grp;
                H8 o;
#pragma unroll
                for (int v = 0; v < 8; ++v) {
                    float conv = (sidx == 0) ? ((t == 0) ? acc00[v] : acc10[v])
                                             : ((t == 0) ? acc01[v] : acc11[v]);
                    float hh = (float)pc[base + v];
                    o.h[v] = (_Float16)(tanhf(0.8f * hh + conv + sBup[base + v]) * fre);
                }
                *(uint4*)(outp + base) = o.u;
            }
        }
    }

    __syncthreads();
    if (tid == 0) {
        atomicAdd(&accum[ACC_PW    + step - 1], sRed[0]);
        atomicAdd(&accum[ACC_DELTA + step - 1], sRed[1]);
        atomicAdd(&accum[ACC_NORM  + step],     sRed[2]);
        if (step == NSTEPS) {
            atomicAdd(&accum[ACC_SAT],  sRed[3]);
            atomicAdd(&accum[ACC_LEAK], sRed[4]);
        }
    }
}

// ---------------------------------------------------------------------------
// finalize: stats + per-target phase logits
// ---------------------------------------------------------------------------
__global__ void finalize_kernel(const float* __restrict__ sRf, const float* __restrict__ sIf,
                                const int* __restrict__ target,
                                const float* __restrict__ accum, float* __restrict__ out)
{
    int t = threadIdx.x;
    if (t < BATCH) {
        int ty = target[2 * t + 0], tx = target[2 * t + 1];
        size_t pidx = ((size_t)t * IMH + ty) * IMW + tx;
        float vr = sRf[pidx], vi = sIf[pidx];
        float mag = sqrtf(vr * vr + vi * vi + 1e-8f);
        out[t * 9 + 0] = (0.35f - mag) * 12.0f;
#pragma unroll
        for (int k = 0; k < 8; ++k) {
            float th = 6.2831853071795864f * (float)k / 8.0f;
            out[t * 9 + 1 + k] = (vr * __cosf(th) + vi * __sinf(th) - 0.35f) * 12.0f;
        }
    } else if (t == BATCH) {
        const float BHW = (float)BATCH * (float)NPIX;
        float dsum = 0.0f;
        for (int s = 0; s < NSTEPS; ++s)
            dsum += sqrtf(accum[ACC_DELTA + s] / (2.0f * BHW));
        float nsum = 0.0f;
        for (int s = 0; s <= NSTEPS; ++s) nsum += accum[ACC_NORM + s];
        float pw = 0.0f;
        for (int s = 0; s < NSTEPS; ++s) pw += accum[ACC_PW + s] / BHW;
        out[72] = dsum / (float)NSTEPS;                       // delta
        out[73] = nsum / ((float)(NSTEPS + 1) * BHW);         // norm_by_step
        out[74] = accum[ACC_SAT] / (2.0f * BHW);              // sat
        out[75] = pw / (float)NSTEPS;                         // pw
        out[76] = accum[ACC_LEAK] / BHW;                      // leak
    }
}

// ---------------------------------------------------------------------------
extern "C" void kernel_launch(void* const* d_in, const int* in_sizes, int n_in,
                              void* d_out, int out_size, void* d_ws, size_t ws_size,
                              hipStream_t stream)
{
    const float* x       = (const float*)d_in[0];
    const int*   target  = (const int*)  d_in[1];
    const float* w_in    = (const float*)d_in[2];
    const float* b_in    = (const float*)d_in[3];
    const float* w_up    = (const float*)d_in[4];
    const float* b_up    = (const float*)d_in[5];
    const float* w_gate  = (const float*)d_in[6];
    const float* b_gate  = (const float*)d_in[7];
    const float* angle   = (const float*)d_in[8];
    const float* decay   = (const float*)d_in[9];
    // d_in[10] = steps (device scalar); NSTEPS=16 hard-coded (graph-capture safe)

    const size_t SRC_BYTES = (size_t)BATCH * PPIX * CPP * 2;   // ~51 MB each
    const size_t S_BYTES   = (size_t)BATCH * NPIX * 4;         // 2 MB each

    char* ws = (char*)d_ws;
    _Float16* srcA = (_Float16*)ws;
    _Float16* srcB = (_Float16*)(ws + SRC_BYTES);
    float* sRa  = (float*)(ws + 2 * SRC_BYTES);
    float* sIa  = (float*)(ws + 2 * SRC_BYTES + 1 * S_BYTES);
    float* sRb  = (float*)(ws + 2 * SRC_BYTES + 2 * S_BYTES);
    float* sIb  = (float*)(ws + 2 * SRC_BYTES + 3 * S_BYTES);
    _Float16* W16g = (_Float16*)(ws + 2 * SRC_BYTES + 4 * S_BYTES);
    float* accum   = (float*)(ws + 2 * SRC_BYTES + 4 * S_BYTES + (size_t)WD * KPAD * 2);

    prep_kernel<<<(WD * KPAD + 255) / 256, 256, 0, stream>>>(w_up, angle, decay,
                                                             W16g, accum);
    border_kernel<<<((unsigned)((size_t)BATCH * PPIX + 255) / 256), 256, 0, stream>>>(srcA, srcB);
    init_kernel<<<(BATCH * NPIX) / 256, 256, 0, stream>>>(x, w_in, b_in,
                                                          srcA, sRa, sIa, accum);

    _Float16* srcBufs[2] = { srcA, srcB };
    float* srBufs[2] = { sRa, sRb };
    float* siBufs[2] = { sIa, sIb };

    for (int t = 1; t <= NSTEPS; ++t) {
        int cur = (t - 1) & 1, nxt = t & 1;
        step_kernel<<<dim3(IMW / TW, IMH / TH, BATCH), 256, SH_TOTAL, stream>>>(
            srcBufs[cur], srcBufs[nxt], srBufs[cur], siBufs[cur],
            srBufs[nxt], siBufs[nxt], W16g, w_gate, b_gate, b_up,
            accum, t);
    }

    // after 16 steps the final state sits in buffer 0 (A)
    finalize_kernel<<<1, 64, 0, stream>>>(sRa, sIa, target, accum, (float*)d_out);
}
